// CudaNormFastWeightLinearTransformerLayer_15891378995474
// MI455X (gfx1250) — matrix operand order repaired and checked by reading, not compiled, not run on hardware
//
#include <hip/hip_runtime.h>
#include <hip/hip_bf16.h>
#include <stdint.h>

#define SLEN   2048
#define BSZ    4
#define DMODEL 1024
#define NHEAD  16
#define DHEAD  64
#define NROWS  (SLEN * BSZ)             // 8192
#define NQKV   (NHEAD * (3 * DHEAD + 1))// 3088
#define LN_EPS 1e-5f
#define OSCALE 0.125f                   // 1/sqrt(64)

typedef __attribute__((ext_vector_type(16))) __bf16 v16bf;
typedef __attribute__((ext_vector_type(8)))  __bf16 v8bf;
typedef __attribute__((ext_vector_type(8)))  float  v8f;

__device__ __forceinline__ uint16_t f2bf(float f) {
  uint32_t u = __float_as_uint(f);
  u += 0x7fffu + ((u >> 16) & 1u);      // round-to-nearest-even
  return (uint16_t)(u >> 16);
}

// A fragment for v_wmma_f32_16x16x32_bf16, wave32 layout:
// lane (half=lane>>4) holds K in [k0+16*half, +16) (contiguous, matches B)
__device__ __forceinline__ v16bf load_afrag(const uint16_t* __restrict__ row,
                                            int k0, int half) {
  v8bf lo = *reinterpret_cast<const v8bf*>(row + k0 + half * 16);
  v8bf hi = *reinterpret_cast<const v8bf*>(row + k0 + half * 16 + 8);
  return __builtin_shufflevector(lo, hi, 0, 1, 2, 3, 4, 5, 6, 7,
                                 8, 9, 10, 11, 12, 13, 14, 15);
}

// ---------------------------------------------------------------- convert
__global__ void k_f32_to_bf16(const float* __restrict__ in,
                              uint16_t* __restrict__ out, int n) {
  int i = blockIdx.x * blockDim.x + threadIdx.x;
  if (i < n) out[i] = f2bf(in[i]);
}

// ---------------------------------------------------------------- bf16 WMMA GEMM
// C[M,N] (f32) = A[M,K] (bf16 row-major) * B[N,K]^T (bf16 row-major weights)
// block = 128 threads = 4 waves; wave tile 32(M) x NSUB*16(N).
template <int NSUB>
__global__ __launch_bounds__(128)
void k_gemm_bf16(const uint16_t* __restrict__ A, const uint16_t* __restrict__ B,
                 float* __restrict__ C, int N, int K, int c0base) {
  const int lane = threadIdx.x & 31;
  const int wv   = threadIdx.x >> 5;
  const int half = lane >> 4;
  const int l16  = lane & 15;
  const int m0   = blockIdx.y * 128 + wv * 32;
  const int c0   = c0base + blockIdx.x * (NSUB * 16);

  const uint16_t* Arow0 = A + (size_t)(m0 + l16) * K;
  const uint16_t* Arow1 = A + (size_t)(m0 + 16 + l16) * K;
  const uint16_t* Bp[NSUB];
#pragma unroll
  for (int j = 0; j < NSUB; ++j)
    Bp[j] = B + (size_t)(c0 + j * 16 + l16) * K + (size_t)half * 16;

  const v8f vzero = {0.f, 0.f, 0.f, 0.f, 0.f, 0.f, 0.f, 0.f};
  v8f acc0[NSUB], acc1[NSUB];
#pragma unroll
  for (int j = 0; j < NSUB; ++j) { acc0[j] = vzero; acc1[j] = vzero; }

  for (int k0 = 0; k0 < K; k0 += 32) {
    v16bf a0 = load_afrag(Arow0, k0, half);
    v16bf a1 = load_afrag(Arow1, k0, half);
#pragma unroll
    for (int j = 0; j < NSUB; ++j) {
      v16bf bf = *reinterpret_cast<const v16bf*>(Bp[j] + k0);
      acc0[j] = __builtin_amdgcn_wmma_f32_16x16x32_bf16(
          false, a0, false, bf, (short)0, acc0[j], false, false);
      acc1[j] = __builtin_amdgcn_wmma_f32_16x16x32_bf16(
          false, a1, false, bf, (short)0, acc1[j], false, false);
    }
  }
#pragma unroll
  for (int j = 0; j < NSUB; ++j) {
    float* Cc = C + c0 + j * 16 + l16;
#pragma unroll
    for (int v = 0; v < 8; ++v) {
      Cc[(size_t)(m0 + v + 8 * half) * N]      = acc0[j][v];
      Cc[(size_t)(m0 + 16 + v + 8 * half) * N] = acc1[j][v];
    }
  }
}

// ---------------------------------------------------------------- activations
__global__ __launch_bounds__(64)
void k_qkv_post(const float* __restrict__ qkvb, float* __restrict__ q,
                float* __restrict__ k, float* __restrict__ v,
                float* __restrict__ beta) {
  const int row = blockIdx.x;            // t*BSZ + b
  const int hd  = blockIdx.y;
  const int d   = threadIdx.x;           // 0..63
  const int t = row >> 2, bb = row & 3;
  const float* src = qkvb + (size_t)row * NQKV + hd * (3 * DHEAD + 1);
  float qr = src[d], kr = src[DHEAD + d], vr = src[2 * DHEAD + d];
  float qe = qr > 0.f ? qr + 1.f : __expf(qr);   // elu(x)+1
  float ke = kr > 0.f ? kr + 1.f : __expf(kr);

  __shared__ float red[4];
  float qs = qe, ks = ke;
#pragma unroll
  for (int m = 16; m > 0; m >>= 1) {
    qs += __shfl_xor(qs, m, 32);
    ks += __shfl_xor(ks, m, 32);
  }
  if ((d & 31) == 0) { red[d >> 5] = qs; red[2 + (d >> 5)] = ks; }
  __syncthreads();
  qs = red[0] + red[1];
  ks = red[2] + red[3];

  size_t base = ((size_t)(bb * NHEAD + hd) * SLEN + t) * DHEAD + d;
  q[base] = qe / qs;
  k[base] = ke / ks;
  v[base] = vr;
  if (d == 0)
    beta[(size_t)(bb * NHEAD + hd) * SLEN + t] =
        1.f / (1.f + __expf(-src[3 * DHEAD]));
}

// ---------------------------------------------------------------- delta-rule recurrence
// One block (256 threads, 8 waves) per (b,h). Thread (i,p): i=row of W (0..63),
// p=quarter (0..3) owning W[i][16p..16p+16). Critical path per step ~= 48 FMAs.
// LDS broadcast buffers double-buffered -> 2 barriers/step.
__global__ __launch_bounds__(256)
void k_delta_rule(const float* __restrict__ q, const float* __restrict__ k,
                  const float* __restrict__ v, const float* __restrict__ beta,
                  uint16_t* __restrict__ lo) {
  const int bh  = blockIdx.x;
  const int bb  = bh >> 4, hd = bh & 15;
  const int tid = threadIdx.x;
  const int i   = tid >> 2;          // W row
  const int p   = tid & 3;           // quarter of the row
  const int j0  = p * 16;
  const int wid = tid >> 5;

  __shared__ float sk[2][DHEAD], sq[2][DHEAD], red[2][16];

  float W[16];
#pragma unroll
  for (int j = 0; j < 16; ++j) W[j] = 0.f;
  float S = 0.f;                     // cumsum of k, element i (replicated per p)

  const size_t base = (size_t)bh * SLEN * DHEAD;
  const float* qp = q + base;
  const float* kp = k + base;
  const float* vp = v + base;
  const float* bp = beta + (size_t)bh * SLEN;

  for (int t = 0; t < SLEN; ++t) {
    const int cur = t & 1;
    float qi = qp[t * DHEAD + i];
    float ki = kp[t * DHEAD + i];
    float vi = vp[t * DHEAD + i];

    // key_denom partial (S before update) and denominator partial (S after);
    // every value appears 4x across the block -> scale by 0.25 after combine.
    float pd = S * ki;
    float Sn = S + ki;
    float dd = Sn * qi;
#pragma unroll
    for (int m = 16; m > 0; m >>= 1) {
      pd += __shfl_xor(pd, m, 32);
      dd += __shfl_xor(dd, m, 32);
    }
    if ((tid & 31) == 0) { red[cur][wid] = pd; red[cur][8 + wid] = dd; }
    if (p == 0) sq[cur][i] = qi;
    S = Sn;
    __syncthreads();

    float pds = 0.f, dds = 0.f;
#pragma unroll
    for (int w = 0; w < 8; ++w) { pds += red[cur][w]; dds += red[cur][8 + w]; }
    float kd  = (t == 0) ? 1.f : 0.25f * pds;
    float den = 0.25f * dds;
    float bt  = bp[t] * kd;
    float kn  = ki / (kd + LN_EPS);
    if (p == 0) sk[cur][i] = kn;
    __syncthreads();

    float vold = 0.f;
#pragma unroll
    for (int jj = 0; jj < 16; ++jj) vold += W[jj] * sk[cur][j0 + jj];
    vold += __shfl_xor(vold, 1, 32);
    vold += __shfl_xor(vold, 2, 32);

    float delta = bt * (vi - vold);
    float outv = 0.f;
#pragma unroll
    for (int jj = 0; jj < 16; ++jj) {
      W[jj] += delta * sk[cur][j0 + jj];
      outv  += W[jj] * sq[cur][j0 + jj];
    }
    outv += __shfl_xor(outv, 1, 32);
    outv += __shfl_xor(outv, 2, 32);

    if (p == 0) {
      float res = OSCALE * outv / (den + LN_EPS);
      lo[(size_t)(t * BSZ + bb) * DMODEL + hd * DHEAD + i] = f2bf(res);
    }
    // no trailing barrier: next step writes the other LDS buffer set
  }
}

// ---------------------------------------------------------------- residual + layernorm
__global__ __launch_bounds__(256)
void k_ln(const float* __restrict__ h, const float* __restrict__ attn,
          const float* __restrict__ g, const float* __restrict__ b,
          float* __restrict__ out) {
  const int r = blockIdx.x;
  const int tid = threadIdx.x;
  float x[4], s = 0.f, s2 = 0.f;
#pragma unroll
  for (int j = 0; j < 4; ++j) {
    int c = tid + j * 256;
    float val = h[(size_t)r * DMODEL + c] + attn[(size_t)r * DMODEL + c];
    x[j] = val; s += val; s2 += val * val;
  }
#pragma unroll
  for (int m = 16; m > 0; m >>= 1) {
    s  += __shfl_xor(s, m, 32);
    s2 += __shfl_xor(s2, m, 32);
  }
  __shared__ float red[16];
  if ((tid & 31) == 0) { red[tid >> 5] = s; red[8 + (tid >> 5)] = s2; }
  __syncthreads();
  s = 0.f; s2 = 0.f;
#pragma unroll
  for (int w = 0; w < 8; ++w) { s += red[w]; s2 += red[8 + w]; }
  float mu  = s * (1.f / DMODEL);
  float var = s2 * (1.f / DMODEL) - mu * mu;
  float rs  = rsqrtf(var + LN_EPS);
#pragma unroll
  for (int j = 0; j < 4; ++j) {
    int c = tid + j * 256;
    out[(size_t)r * DMODEL + c] = (x[j] - mu) * rs * g[c] + b[c];
  }
}

// ---------------------------------------------------------------- launch
extern "C" void kernel_launch(void* const* d_in, const int* in_sizes, int n_in,
                              void* d_out, int out_size, void* d_ws, size_t ws_size,
                              hipStream_t stream) {
  const float* h      = (const float*)d_in[0];
  const float* w_qkvb = (const float*)d_in[1];
  const float* w_o    = (const float*)d_in[2];
  const float* ln_g   = (const float*)d_in[3];
  const float* ln_b   = (const float*)d_in[4];
  float* out = (float*)d_out;
  char* ws = (char*)d_ws;

  const size_t o_hbf  = 0;
  const size_t o_wq   = o_hbf + (size_t)NROWS * DMODEL * 2;
  const size_t o_wo   = o_wq  + (size_t)NQKV  * DMODEL * 2;
  const size_t o_qkvb = o_wo  + (size_t)DMODEL * DMODEL * 2;
  const size_t o_lo   = o_qkvb;                                   // reuse after post
  const size_t o_attn = o_qkvb + (size_t)NROWS * DMODEL * 2;      // inside qkvb region
  const size_t o_q    = o_qkvb + (size_t)NROWS * NQKV * 4;
  const size_t o_k    = o_q + (size_t)NROWS * DMODEL * 4;
  const size_t o_v    = o_k + (size_t)NROWS * DMODEL * 4;
  const size_t o_beta = o_v + (size_t)NROWS * DMODEL * 4;

  uint16_t* hbf   = (uint16_t*)(ws + o_hbf);
  uint16_t* wqbf  = (uint16_t*)(ws + o_wq);
  uint16_t* wobf  = (uint16_t*)(ws + o_wo);
  float*    qkvb  = (float*)(ws + o_qkvb);
  uint16_t* lobf  = (uint16_t*)(ws + o_lo);
  float*    attn  = (float*)(ws + o_attn);
  float*    qb    = (float*)(ws + o_q);
  float*    kb    = (float*)(ws + o_k);
  float*    vb    = (float*)(ws + o_v);
  float*    betab = (float*)(ws + o_beta);

  k_f32_to_bf16<<<(NROWS * DMODEL + 255) / 256, 256, 0, stream>>>(h, hbf, NROWS * DMODEL);
  k_f32_to_bf16<<<(NQKV * DMODEL + 255) / 256, 256, 0, stream>>>(w_qkvb, wqbf, NQKV * DMODEL);
  k_f32_to_bf16<<<(DMODEL * DMODEL + 255) / 256, 256, 0, stream>>>(w_o, wobf, DMODEL * DMODEL);

  // QKV projection: 48 full 64-wide column groups + one 16-col remainder.
  k_gemm_bf16<4><<<dim3(48, NROWS / 128), 128, 0, stream>>>(
      hbf, wqbf, qkvb, NQKV, DMODEL, 0);
  k_gemm_bf16<1><<<dim3(1, NROWS / 128), 128, 0, stream>>>(
      hbf, wqbf, qkvb, NQKV, DMODEL, 3072);

  k_qkv_post<<<dim3(NROWS, NHEAD), 64, 0, stream>>>(qkvb, qb, kb, vb, betab);

  k_delta_rule<<<BSZ * NHEAD, 256, 0, stream>>>(qb, kb, vb, betab, lobf);

  // Output projection: N=1024, all full tiles.
  k_gemm_bf16<4><<<dim3(DMODEL / 64, NROWS / 128), 128, 0, stream>>>(
      lobf, wobf, attn, DMODEL, DMODEL, 0);

  k_ln<<<NROWS, 256, 0, stream>>>(h, attn, ln_g, ln_b, out);
}
